// encoder_77051713290942
// MI455X (gfx1250) — matrix operand order repaired and checked by reading, not compiled, not run on hardware
//
#include <hip/hip_runtime.h>
#include <math.h>

#define LH     300      // hidden size
#define FOURL  1200     // 4*H gate rows
#define TSEQ   8192     // sequence length
#define KP     320      // K padded to 10 tiles of 32 for f16 WMMA
#define NWG    25       // workgroups in the scan (300 units / 12)
#define UPW    12       // hidden units per workgroup
#define ROWS   48       // gate rows per workgroup (UPW*4)
#define WSTR   301      // LDS row stride (odd -> conflict-free across wave32)
#define SCAN_T 64       // scan block size (2 waves)

typedef __attribute__((ext_vector_type(8)))  _Float16 h8;
typedef __attribute__((ext_vector_type(16))) _Float16 v16h;
typedef __attribute__((ext_vector_type(8)))  float    v8f;

// ---------------------------------------------------------------------------
// Prep: fp32 -> f16 with zero-pad K 300->320, bias fold, h-buffer/counter init
// ---------------------------------------------------------------------------
__global__ void prep_kernel(const float* __restrict__ x,
                            const float* __restrict__ Wih,
                            const float* __restrict__ bih,
                            const float* __restrict__ bhh,
                            const float* __restrict__ h0,
                            _Float16* __restrict__ xh,
                            _Float16* __restrict__ wihh,
                            float* __restrict__ bias,
                            float* __restrict__ hbuf,
                            unsigned* __restrict__ cnt) {
  long i = (long)blockIdx.x * blockDim.x + threadIdx.x;
  if (i < (long)TSEQ * KP) {
    int r = (int)(i / KP), k = (int)(i % KP);
    xh[i] = (k < LH) ? (_Float16)x[(size_t)r * LH + k] : (_Float16)0.f;
  }
  if (i < (long)FOURL * KP) {
    int r = (int)(i / KP), k = (int)(i % KP);
    wihh[i] = (k < LH) ? (_Float16)Wih[(size_t)r * LH + k] : (_Float16)0.f;
  }
  if (i < FOURL) bias[i] = bih[i] + bhh[i];
  if (i < LH)   { hbuf[i] = h0[i]; hbuf[LH + i] = 0.f; }
  if (i == 0)   *cnt = 0u;
}

// ---------------------------------------------------------------------------
// x_proj = x @ W_ih^T + (b_ih + b_hh), one 16x16 tile per wave via WMMA f16
// ---------------------------------------------------------------------------
__global__ void __launch_bounds__(32)
xproj_wmma_kernel(const _Float16* __restrict__ xh,
                  const _Float16* __restrict__ wihh,
                  const float* __restrict__ bias,
                  float* __restrict__ xproj) {
  const int lane  = threadIdx.x & 31;
  const int lhi   = lane >> 4;          // 0: lanes 0-15, 1: lanes 16-31
  const int lmod  = lane & 15;
  const int r0    = blockIdx.x * 16;    // row tile in [0,8192)
  const int c0    = blockIdx.y * 16;    // col tile in [0,1200)

  // A: row r0+lmod, K chunks at +{0,8} (lanes<16) or +{16,24} (lanes>=16)
  const _Float16* aBase = xh   + (size_t)(r0 + lmod) * KP + (lhi ? 16 : 0);
  // B[k][n] = W_ih[n][k]: row c0+lmod of wihh, K chunk at +0 or +16
  const _Float16* bBase = wihh + (size_t)(c0 + lmod) * KP + (lhi ? 16 : 0);

  v8f acc = {};
#pragma unroll
  for (int kt = 0; kt < KP; kt += 32) {
    h8 alo = *(const h8*)(aBase + kt);
    h8 ahi = *(const h8*)(aBase + kt + 8);
    v16h a = __builtin_shufflevector(alo, ahi,
                                     0,1,2,3,4,5,6,7,8,9,10,11,12,13,14,15);
    v16h b = *(const v16h*)(bBase + kt);
    acc = __builtin_amdgcn_wmma_f32_16x16x32_f16(
        /*neg_a=*/false, a, /*neg_b=*/false, b,
        /*c_mod=*/(short)0, acc, /*reuse_a=*/false, /*reuse_b=*/false);
  }

  const float bn = bias[c0 + lmod];
#pragma unroll
  for (int v = 0; v < 8; ++v) {
    int m = v + (lhi ? 8 : 0);
    xproj[(size_t)(r0 + m) * FOURL + (c0 + lmod)] = acc[v] + bn;
  }
}

// ---------------------------------------------------------------------------
// Sequential LSTM scan: 25 WGs, W_hh slice LDS-resident, h double-buffered in
// global, device-scope counter barrier per step.
// ---------------------------------------------------------------------------
__global__ void __launch_bounds__(SCAN_T)
lstm_scan_kernel(const float* __restrict__ Whh,
                 const float* __restrict__ c0,
                 const float* __restrict__ xproj,
                 float* __restrict__ hbuf,       // [2][LH]
                 unsigned* __restrict__ cnt,
                 float* __restrict__ out) {      // [h(300), c(300)]
  __shared__ float w[ROWS * WSTR];
  __shared__ float hl[LH];
  __shared__ float gl[ROWS];

  const int t  = threadIdx.x;
  const int u0 = blockIdx.x * UPW;

  // Stage this WG's W_hh slice into LDS: LDS row (u*4+g) <- W_hh[g*LH + u0+u]
  for (int idx = t; idx < ROWS * LH; idx += SCAN_T) {
    int row = idx / LH, k = idx - row * LH;
    int u = row >> 2, g = row & 3;
    w[row * WSTR + k] = Whh[(size_t)(g * LH + u0 + u) * LH + k];
  }
  for (int k = t; k < LH; k += SCAN_T) hl[k] = hbuf[k];   // h for step 0

  float cval = (t < UPW) ? c0[u0 + t] : 0.f;
  float hval = 0.f;
  __syncthreads();

  for (int s = 0; s < TSEQ; ++s) {
    const int wb = (s & 1) ^ 1;  // write buffer; read buffer = s&1 (== hl)

    if (s + 1 < TSEQ)
      __builtin_prefetch(xproj + (size_t)(s + 1) * FOURL + u0, 0, 1);

    if (t < ROWS) {
      const float* wr = w + t * WSTR;
      float acc = 0.f;
#pragma unroll 4
      for (int k = 0; k < LH; ++k) acc = fmaf(wr[k], hl[k], acc);
      const int u = t >> 2, g = t & 3;
      acc += xproj[(size_t)s * FOURL + g * LH + u0 + u];
      gl[t] = acc;
    }
    __syncthreads();

    if (t < UPW) {
      float gi = gl[t * 4 + 0], gf = gl[t * 4 + 1];
      float gg = gl[t * 4 + 2], go = gl[t * 4 + 3];
      gi = 1.f / (1.f + __expf(-gi));
      gf = 1.f / (1.f + __expf(-gf));
      gg = tanhf(gg);
      go = 1.f / (1.f + __expf(-go));
      cval = gf * cval + gi * gg;
      hval = go * tanhf(cval);
      hbuf[wb * LH + u0 + t] = hval;
      __threadfence();                       // release h slice
    }
    __syncthreads();

    if (t == 0) {                            // grid-wide step barrier
      atomicAdd(cnt, 1u);
      const unsigned target = (unsigned)NWG * (unsigned)(s + 1);
      while (__hip_atomic_load(cnt, __ATOMIC_RELAXED,
                               __HIP_MEMORY_SCOPE_AGENT) < target)
        __builtin_amdgcn_s_sleep(1);
      __threadfence();                       // acquire other WGs' h slices
    }
    __syncthreads();

    for (int k = t; k < LH; k += SCAN_T) hl[k] = hbuf[wb * LH + k];
    __syncthreads();
  }

  if (t < UPW) {
    out[u0 + t]      = hval;   // final h
    out[LH + u0 + t] = cval;   // final c
  }
}

// ---------------------------------------------------------------------------
extern "C" void kernel_launch(void* const* d_in, const int* in_sizes, int n_in,
                              void* d_out, int out_size, void* d_ws, size_t ws_size,
                              hipStream_t stream) {
  const float* x   = (const float*)d_in[0];
  const float* h0  = (const float*)d_in[1];
  const float* c0  = (const float*)d_in[2];
  const float* Wih = (const float*)d_in[3];
  const float* Whh = (const float*)d_in[4];
  const float* bih = (const float*)d_in[5];
  const float* bhh = (const float*)d_in[6];
  float* out = (float*)d_out;
  (void)in_sizes; (void)n_in; (void)out_size;

  char* ws = (char*)d_ws;
  size_t off = 0;
  auto carve = [&](size_t bytes) -> char* {
    off = (off + 511) & ~(size_t)511;
    char* p = ws + off;
    off += bytes;
    return p;
  };
  _Float16* xh   = (_Float16*)carve((size_t)TSEQ * KP * sizeof(_Float16));
  _Float16* wihh = (_Float16*)carve((size_t)FOURL * KP * sizeof(_Float16));
  float*    bias = (float*)   carve((size_t)FOURL * sizeof(float));
  float*    xprj = (float*)   carve((size_t)TSEQ * FOURL * sizeof(float));
  float*    hbuf = (float*)   carve((size_t)2 * LH * sizeof(float));
  unsigned* cnt  = (unsigned*)carve(64);
  if (off > ws_size) return;  // scratch too small; bail deterministically

  {
    long n = (long)TSEQ * KP;
    unsigned nb = (unsigned)((n + 255) / 256);
    prep_kernel<<<dim3(nb), dim3(256), 0, stream>>>(
        x, Wih, bih, bhh, h0, xh, wihh, bias, hbuf, cnt);
  }
  xproj_wmma_kernel<<<dim3(TSEQ / 16, FOURL / 16), dim3(32), 0, stream>>>(
      xh, wihh, bias, xprj);
  lstm_scan_kernel<<<dim3(NWG), dim3(SCAN_T), 0, stream>>>(
      Whh, c0, xprj, hbuf, cnt, out);
}
